// MemoryBank_6992206758350
// MI455X (gfx1250) — compile-verified, matrix operand order checked by reading, not run on hardware
//
#include <hip/hip_runtime.h>
#include <hip/hip_bf16.h>
#include <math.h>

typedef __attribute__((ext_vector_type(16))) __bf16 v16bf;
typedef __attribute__((ext_vector_type(8)))  float  v8f;

#define H_DIM   1024
#define ROWS    64      // rows per block (4 WMMA row tiles)
#define CHUNK   128     // keys per outer iteration (8 WMMA col tiles)
#define HBLK    128     // H elements staged per K-tile load
#define TOPK    32
#define THREADS 256

union FragU { uint4 u[2]; v16bf v; };

__device__ __forceinline__ unsigned bfpack2(float a, float b) {
  unsigned ua = __float_as_uint(a), ub = __float_as_uint(b);
  unsigned ra = (ua + 0x7FFFu + ((ua >> 16) & 1u)) >> 16;   // round-to-nearest-even
  unsigned rb = (ub + 0x7FFFu + ((ub >> 16) & 1u)) >> 16;
  return (ra & 0xFFFFu) | (rb << 16);
}

// 16B global -> LDS async copy (ASYNCcnt-tracked, bypasses VGPRs)
__device__ __forceinline__ void async_copy16(unsigned lds_byte, const void* gaddr) {
  asm volatile("global_load_async_to_lds_b128 %0, %1, off"
               :: "v"(lds_byte), "v"(gaddr)
               : "memory");
}
__device__ __forceinline__ void wait_async0() {
  asm volatile("s_wait_asynccnt 0x0" ::: "memory");
}

// stage CHUNK x HBLK bf16 keys from pre-converted K into LDS via async DMA
__device__ __forceinline__ void stage_ktile_async(const unsigned short* __restrict__ kbf,
                                                  unsigned ktile_lds_byte,
                                                  int csup, int hblk, int tid) {
  // 32KB = 2048 x 16B units; 8 units per thread
  #pragma unroll
  for (int i = 0; i < (CHUNK * HBLK * 2 / 16) / THREADS; ++i) {   // 8 iters
    int u   = tid + i * THREADS;
    int key = u >> 4;                   // 16 x 16B = 256B per key row
    int off = (u & 15) * 16;
    const void* g = (const void*)((const char*)kbf
                     + ((long)(csup * CHUNK + key) * H_DIM + hblk * HBLK) * 2 + off);
    async_copy16(ktile_lds_byte + key * (HBLK * 2) + off, g);
  }
}

// ---------------------------------------------------------------------------
// Kernel 0: one-time K fp32 -> bf16 conversion into workspace (HBM-bound, ~5us)
// ---------------------------------------------------------------------------
__global__ void __launch_bounds__(256)
membank_cvt_keys(const float* __restrict__ keys, unsigned* __restrict__ kbf, long n4) {
  long i = (long)blockIdx.x * 256 + threadIdx.x;
  if (i < n4) {
    const float4 f = ((const float4*)keys)[i];
    kbf[i * 2 + 0] = bfpack2(f.x, f.y);
    kbf[i * 2 + 1] = bfpack2(f.z, f.w);
  }
}

// ---------------------------------------------------------------------------
// Phase 1: fused  scores = Q*K^T/sqrt(H)  +  running top-32 per row.
// 256 threads (8 waves) = 2 row-groups x 4 col-groups; each wave 2x2 tiles
// (2:1 LDS:WMMA).  K stage double-buffered via async global->LDS DMA.
// ---------------------------------------------------------------------------
__global__ void __launch_bounds__(THREADS)
membank_scores_topk(const float* __restrict__ q,
                    const unsigned short* __restrict__ kbf,   // pre-converted bf16 keys
                    float* __restrict__ attn_out,   // [rows, 32] weights (descending)
                    int*   __restrict__ idx_out,    // [rows, 32] key indices
                    int M) {
  extern __shared__ char smem[];
  unsigned short* qtile  = (unsigned short*)smem;                         // ROWS x H bf16 (128KB)
  unsigned short* ktileA = (unsigned short*)(smem + ROWS*H_DIM*2);        // CHUNK x HBLK (32KB)
  unsigned short* ktileB = ktileA + CHUNK*HBLK;                           // (32KB)
  float* stile = (float*)(smem + ROWS*H_DIM*2 + 2*CHUNK*HBLK*2);          // ROWS x CHUNK (32KB)
  float* topv  = (float*)((char*)stile + ROWS*CHUNK*4);                   // ROWS x 32 (8KB)
  int*   topi  = (int*)((char*)topv + ROWS*TOPK*4);                       // ROWS x 32 (8KB)

  // low 32 bits of a generic shared pointer == LDS byte address
  const unsigned ldsA = (unsigned)(uintptr_t)(void*)ktileA;
  const unsigned ldsB = (unsigned)(uintptr_t)(void*)ktileB;

  const int tid  = threadIdx.x;
  const int lane = tid & 31;
  const int wave = tid >> 5;
  const int rg   = wave & 1;          // row group: row tiles {2rg, 2rg+1}
  const int cg   = wave >> 1;         // col group: col tiles {2cg, 2cg+1}
  const int nl   = lane & 15;
  const int hi   = lane >> 4;
  const long rowbase = (long)blockIdx.x * ROWS;

  // ---- stage Q tile: ROWS x H, fp32 -> bf16 in LDS (done once)
  for (int i = 0; i < (ROWS * H_DIM / 4) / THREADS; ++i) {   // 64 iters
    int idx4 = tid + i * THREADS;
    int flat = idx4 * 4;
    int row  = flat >> 10;            // / H_DIM
    int h    = flat & (H_DIM - 1);
    const float4 f = *(const float4*)(q + (rowbase + row) * H_DIM + h);
    ((unsigned*)qtile)[idx4 * 2 + 0] = bfpack2(f.x, f.y);
    ((unsigned*)qtile)[idx4 * 2 + 1] = bfpack2(f.z, f.w);
  }

  // ---- init per-row top-k state
  float minv = -INFINITY; int minpos = 0;
  if (tid < ROWS) {
    for (int j = 0; j < TOPK; ++j) { topv[tid*TOPK + j] = -INFINITY; topi[tid*TOPK + j] = 0; }
  }
  __syncthreads();

  const float inv_scale = 0.03125f;   // 1/sqrt(1024)

  for (int csup = 0; csup < M / CHUNK; ++csup) {
    v8f acc00 = {0,0,0,0,0,0,0,0};
    v8f acc01 = {0,0,0,0,0,0,0,0};
    v8f acc10 = {0,0,0,0,0,0,0,0};
    v8f acc11 = {0,0,0,0,0,0,0,0};

    // warm L2 for the next key super-tile (speculative, OOB-safe)
    if (csup + 1 < M / CHUNK)
      __builtin_prefetch(kbf + ((long)(csup + 1) * CHUNK + (tid & (CHUNK-1))) * H_DIM, 0, 0);

    // prologue: stage first K tile
    stage_ktile_async(kbf, ldsA, csup, 0, tid);
    wait_async0();
    __syncthreads();

    for (int hblk = 0; hblk < H_DIM / HBLK; ++hblk) {
      const unsigned short* cur = (hblk & 1) ? ktileB : ktileA;
      const unsigned        nxt = (hblk & 1) ? ldsA   : ldsB;

      // stage next tile via async DMA while computing current
      if (hblk + 1 < H_DIM / HBLK)
        stage_ktile_async(kbf, nxt, csup, hblk + 1, tid);

      // 4 k-chunks of 32 -> 16 WMMAs per wave per hblk (2x2 blocking)
      #pragma unroll
      for (int kc = 0; kc < HBLK / 32; ++kc) {
        FragU a0, a1, b0, b1;
        // A fragments (ISA 16-bit A layout): lane holds its row,
        // two 16B chunks at k = hi*8 and k = 16 + hi*8 within the 32-chunk.
        const unsigned short* qp0 = qtile + (rg*32 + nl) * H_DIM + hblk*HBLK + kc*32 + hi*8;
        a0.u[0] = *(const uint4*)(qp0);
        a0.u[1] = *(const uint4*)(qp0 + 16);
        const unsigned short* qp1 = qp0 + 16 * H_DIM;
        a1.u[0] = *(const uint4*)(qp1);
        a1.u[1] = *(const uint4*)(qp1 + 16);
        // B fragments: lane holds its key column, same k pattern.
        const unsigned short* kp0 = cur + (cg*32 + nl) * HBLK + kc*32 + hi*8;
        b0.u[0] = *(const uint4*)(kp0);
        b0.u[1] = *(const uint4*)(kp0 + 16);
        const unsigned short* kp1 = kp0 + 16 * HBLK;
        b1.u[0] = *(const uint4*)(kp1);
        b1.u[1] = *(const uint4*)(kp1 + 16);
        acc00 = __builtin_amdgcn_wmma_f32_16x16x32_bf16(false, a0.v, false, b0.v,
                                                        (short)0, acc00, false, false);
        acc01 = __builtin_amdgcn_wmma_f32_16x16x32_bf16(false, a0.v, false, b1.v,
                                                        (short)0, acc01, false, false);
        acc10 = __builtin_amdgcn_wmma_f32_16x16x32_bf16(false, a1.v, false, b0.v,
                                                        (short)0, acc10, false, false);
        acc11 = __builtin_amdgcn_wmma_f32_16x16x32_bf16(false, a1.v, false, b1.v,
                                                        (short)0, acc11, false, false);
      }
      // own async stores done, then barrier: next buffer staged by all AND
      // current buffer consumed by all
      wait_async0();
      __syncthreads();
    }

    // scatter scores to LDS (C layout: element v -> row hi*8+v, col nl)
    #pragma unroll
    for (int v = 0; v < 8; ++v) {
      int m = hi * 8 + v;
      stile[(rg*32 +      m) * CHUNK + cg*32 + nl]      = acc00[v] * inv_scale;
      stile[(rg*32 +      m) * CHUNK + cg*32 + 16 + nl] = acc01[v] * inv_scale;
      stile[(rg*32 + 16 + m) * CHUNK + cg*32 + nl]      = acc10[v] * inv_scale;
      stile[(rg*32 + 16 + m) * CHUNK + cg*32 + 16 + nl] = acc11[v] * inv_scale;
    }
    __syncthreads();

    // per-row running top-32 (thread tid owns row tid)
    if (tid < ROWS) {
      const int base = csup * CHUNK;
      float* tv = topv + tid * TOPK;
      int*   ti = topi + tid * TOPK;
      for (int c = 0; c < CHUNK; ++c) {
        float s = stile[tid * CHUNK + c];
        if (s > minv) {
          tv[minpos] = s; ti[minpos] = base + c;
          float mv = INFINITY; int mp = 0;
          for (int j = 0; j < TOPK; ++j) { float t = tv[j]; if (t < mv) { mv = t; mp = j; } }
          minv = mv; minpos = mp;
        }
      }
    }
    __syncthreads();
  }

  // ---- finalize: sort descending, softmax, emit weights + indices
  if (tid < ROWS) {
    float* tv = topv + tid * TOPK;
    int*   ti = topi + tid * TOPK;
    for (int a2 = 0; a2 < TOPK - 1; ++a2) {
      int best = a2; float bv = tv[a2];
      for (int j = a2 + 1; j < TOPK; ++j) { float t = tv[j]; if (t > bv) { bv = t; best = j; } }
      float tmpv = tv[a2]; tv[a2] = tv[best]; tv[best] = tmpv;
      int   tmpi = ti[a2]; ti[a2] = ti[best]; ti[best] = tmpi;
    }
    float m0 = tv[0];
    float sum = 0.f;
    for (int j = 0; j < TOPK; ++j) sum += __expf(tv[j] - m0);
    float inv = 1.f / sum;
    long rowg = rowbase + tid;
    for (int j = 0; j < TOPK; ++j) {
      attn_out[rowg * TOPK + j] = __expf(tv[j] - m0) * inv;
      idx_out[rowg * TOPK + j]  = ti[j];
    }
  }
}

// ---------------------------------------------------------------------------
// Phase 2: out[row] = sigmoid(q·gw + gb) * sum_k w_k * V[idx_k].  1 block/row.
// ---------------------------------------------------------------------------
__global__ void __launch_bounds__(256)
membank_gather(const float* __restrict__ q,
               const float* __restrict__ vals,
               const float* __restrict__ gw,
               const float* __restrict__ gb,
               const float* __restrict__ attn,
               const int*   __restrict__ idxw,
               float* __restrict__ out) {
  __shared__ float wsh[TOPK];
  __shared__ int   ish[TOPK];
  __shared__ float red[256];
  const long row = blockIdx.x;
  const int  tid = threadIdx.x;
  const int  h0  = tid * 4;

  float4 qv = *(const float4*)(q + row * H_DIM + h0);
  float4 g4 = *(const float4*)(gw + h0);
  red[tid] = qv.x*g4.x + qv.y*g4.y + qv.z*g4.z + qv.w*g4.w;
  if (tid < TOPK) {
    wsh[tid] = attn[row * TOPK + tid];
    ish[tid] = idxw[row * TOPK + tid];
  }
  __syncthreads();
  for (int s = 128; s > 0; s >>= 1) {
    if (tid < s) red[tid] += red[tid + s];
    __syncthreads();
  }
  float gate = 1.f / (1.f + __expf(-(red[0] + gb[0])));

  float4 acc = {0.f, 0.f, 0.f, 0.f};
  #pragma unroll 8
  for (int k = 0; k < TOPK; ++k) {
    const float4 v = *(const float4*)(vals + (long)ish[k] * H_DIM + h0);
    float wk = wsh[k];
    acc.x += wk * v.x; acc.y += wk * v.y; acc.z += wk * v.z; acc.w += wk * v.w;
  }
  acc.x *= gate; acc.y *= gate; acc.z *= gate; acc.w *= gate;
  *(float4*)(out + row * H_DIM + h0) = acc;
}

// ---------------------------------------------------------------------------
extern "C" void kernel_launch(void* const* d_in, const int* in_sizes, int n_in,
                              void* d_out, int out_size, void* d_ws, size_t ws_size,
                              hipStream_t stream) {
  (void)n_in; (void)out_size; (void)ws_size;
  const float* q    = (const float*)d_in[0];
  const float* keys = (const float*)d_in[1];
  const float* vals = (const float*)d_in[2];
  const float* gw   = (const float*)d_in[3];
  const float* gb   = (const float*)d_in[4];
  // d_in[5] is top_k; fixed to 32 per setup_inputs (can't read device scalar in capture).

  const int  H    = in_sizes[3];                    // 1024
  const long rows = (long)in_sizes[0] / H;          // 8192
  const int  M    = in_sizes[1] / H;                // 16384

  float* out_mem = (float*)d_out;
  float* attn    = out_mem + rows * (long)H;        // second output, concatenated

  // workspace: [K bf16: M*H*2 bytes][top-k indices: rows*32*4 bytes] = 33 MB
  unsigned short* kbf = (unsigned short*)d_ws;
  int* idxw = (int*)((char*)d_ws + (size_t)M * H * 2);

  // kernel 0: K fp32 -> bf16 (HBM-bound, ~5us; result L2-resident thereafter)
  long n4 = (long)M * H / 4;
  membank_cvt_keys<<<(int)((n4 + 255) / 256), 256, 0, stream>>>(keys, (unsigned*)kbf, n4);

  size_t smem = (size_t)ROWS*H_DIM*2 + (size_t)2*CHUNK*HBLK*2
              + (size_t)ROWS*CHUNK*4 + (size_t)ROWS*TOPK*8;   // 240 KB
  hipFuncSetAttribute((const void*)membank_scores_topk,
                      hipFuncAttributeMaxDynamicSharedMemorySize, (int)smem);

  membank_scores_topk<<<(int)(rows / ROWS), THREADS, smem, stream>>>(q, kbf, attn, idxw, M);
  membank_gather<<<(int)rows, 256, 0, stream>>>(q, vals, gw, gb, attn, idxw, out_mem);
}
